// LogicRecursiveNN_24068996726977
// MI455X (gfx1250) — matrix operand order repaired and checked by reading, not compiled, not run on hardware
//
#include <hip/hip_runtime.h>
#include <hip/hip_bf16.h>

typedef __bf16 bf16_t;
typedef bf16_t v16bf __attribute__((ext_vector_type(16)));
typedef bf16_t v8bf  __attribute__((ext_vector_type(8)));
typedef float  v8f   __attribute__((ext_vector_type(8)));

// ---------------------------------------------------------------------------
// Weight convert + transpose: in f32 [K][N] row-major -> out bf16 [N][K]
// ---------------------------------------------------------------------------
__global__ void cvt_transpose_kernel(const float* __restrict__ in,
                                     bf16_t* __restrict__ out, int K, int N) {
  int i = blockIdx.x * 256 + threadIdx.x;
  if (i >= K * N) return;
  int k = i / N, n = i % N;
  out[(size_t)n * K + k] = (bf16_t)in[i];
}

// ---------------------------------------------------------------------------
// Gather leaf embeddings, f32 -> bf16. One row (E=512) per block.
// ---------------------------------------------------------------------------
__global__ void gather_emb_kernel(const int* __restrict__ idx,
                                  const float* __restrict__ emb,
                                  bf16_t* __restrict__ out) {
  int row = blockIdx.x;
  int e = idx[row];
  const float* src = emb + (size_t)e * 512;
  bf16_t* dst = out + (size_t)row * 512;
  for (int c = threadIdx.x; c < 512; c += 256) dst[c] = (bf16_t)src[c];
}

// ---------------------------------------------------------------------------
// feat[b, 0:512]   = th_emb
// feat[b, 512:1536]= root viewed as [512, 1024] (contiguous)
// ---------------------------------------------------------------------------
__global__ void build_feat_kernel(const float* __restrict__ th,
                                  const bf16_t* __restrict__ root,
                                  bf16_t* __restrict__ feat) {
  int i = blockIdx.x * 256 + threadIdx.x;
  if (i >= 512 * 1536) return;
  int b = i / 1536, c = i % 1536;
  feat[i] = (c < 512) ? (bf16_t)th[c] : root[(size_t)b * 1024 + (c - 512)];
}

// ---------------------------------------------------------------------------
// Final layer: out[b] = sigmoid(dot(z3[b, :128], h4) + hb4). One wave per row.
// ---------------------------------------------------------------------------
__global__ void head4_kernel(const bf16_t* __restrict__ z3,
                             const float* __restrict__ w,
                             const float* __restrict__ bias,
                             float* __restrict__ out) {
  int b = blockIdx.x;
  int lane = threadIdx.x;  // blockDim.x == 32 (one wave32)
  float s = 0.f;
  for (int k = lane; k < 128; k += 32) s += (float)z3[(size_t)b * 128 + k] * w[k];
  for (int off = 16; off > 0; off >>= 1) s += __shfl_down(s, off, 32);
  if (lane == 0) out[b] = 1.f / (1.f + __expf(-(s + bias[0])));
}

// ---------------------------------------------------------------------------
// bf16 WMMA GEMM: C[M,N] = act(A[M,K] @ Bt[N,K]^T + bias[N])
//   A row-major bf16, Bt = B transposed (N rows of K) bf16, C bf16 row-major.
//   128x128 tile per 256-thread workgroup (8 waves, wave32).
//   Wave grid 4(M) x 2(N): each wave -> 32x64 = 2x4 v_wmma_f32_16x16x32_bf16.
//   Global->LDS staging uses CDNA5 async copies (global_load_async_to_lds_b128,
//   ASYNCcnt) with LDS double buffering: prefetch tile i+1 while computing i.
//   Requires: M % 128 == 0, N % 128 == 0, K % 32 == 0.
//   ACT: 0 = ReLU, 1 = tanh
// ---------------------------------------------------------------------------
template <int ACT>
__global__ __launch_bounds__(256) void gemm_bf16_kernel(
    const bf16_t* __restrict__ A, const bf16_t* __restrict__ Bt,
    const float* __restrict__ bias, bf16_t* __restrict__ C,
    int M, int N, int K) {
  __shared__ bf16_t Al[2][128][40];  // 32-wide K slab, +8 pad (16B align kept)
  __shared__ bf16_t Bl[2][128][40];  // rows are N-indices of Bt

  const int tid    = threadIdx.x;
  const int lane   = tid & 31;
  const int wave   = tid >> 5;
  const int waveM  = wave >> 1;   // 0..3
  const int waveN  = wave & 1;    // 0..1
  const int laneLo = lane & 15;
  const int hi     = lane >> 4;   // 0 or 1 (lane group)
  const int rowBase = blockIdx.y * 128;
  const int colBase = blockIdx.x * 128;

  // Per-thread staging coordinates (2 x 16B for A, 2 x 16B for B per tile).
  // 4 async-load instructions per wave per tile => ASYNCcnt thresholds 4 / 0.
  auto stage = [&](int buf, int k0) {
#pragma unroll
    for (int p = 0; p < 2; ++p) {
      int idx = p * 256 + tid;          // 0..511
      int r   = idx >> 2;               // 0..127
      int cg  = (idx & 3) * 8;          // 0,8,16,24 halfs
      unsigned alds = (unsigned)(size_t)&Al[buf][r][cg];  // low 32b = LDS offset
      unsigned blds = (unsigned)(size_t)&Bl[buf][r][cg];
      const bf16_t* ag = A  + (size_t)(rowBase + r) * K + k0 + cg;
      const bf16_t* bg = Bt + (size_t)(colBase + r) * K + k0 + cg;
      asm volatile("global_load_async_to_lds_b128 %0, %1, off"
                   :: "v"(alds), "v"(ag) : "memory");
      asm volatile("global_load_async_to_lds_b128 %0, %1, off"
                   :: "v"(blds), "v"(bg) : "memory");
    }
  };

  v8f acc[2][4];
#pragma unroll
  for (int mf = 0; mf < 2; ++mf)
#pragma unroll
    for (int nf = 0; nf < 4; ++nf) {
      v8f z = {0.f, 0.f, 0.f, 0.f, 0.f, 0.f, 0.f, 0.f};
      acc[mf][nf] = z;
    }

  const int nk = K >> 5;  // K/32 steps
  stage(0, 0);
  for (int i = 0; i < nk; ++i) {
    const int cur = i & 1;
    if (i + 1 < nk) {
      stage(cur ^ 1, (i + 1) * 32);           // prefetch next tile
      asm volatile("s_wait_asynccnt 4" ::: "memory");  // tile i resident
    } else {
      asm volatile("s_wait_asynccnt 0" ::: "memory");
    }
    __syncthreads();  // publish async LDS writes to all 8 waves

    // ---- build fragments per ISA 16-bit A/B VGPR layouts ----
    v16bf afrag[2], bfrag[4];
#pragma unroll
    for (int mf = 0; mf < 2; ++mf) {
      int r = waveM * 32 + mf * 16 + laneLo;
      // lanes 0-15: elems 0-7 = K0..7, 8-15 = K16..23 ; lanes 16-31: +8
      v8bf a0 = *(const v8bf*)&Al[cur][r][hi * 8];
      v8bf a1 = *(const v8bf*)&Al[cur][r][16 + hi * 8];
      afrag[mf] = __builtin_shufflevector(a0, a1, 0, 1, 2, 3, 4, 5, 6, 7,
                                          8, 9, 10, 11, 12, 13, 14, 15);
    }
#pragma unroll
    for (int nf = 0; nf < 4; ++nf) {
      int r = waveN * 64 + nf * 16 + laneLo;
      // lanes 0-15: elems 0-7 = K0..7, 8-15 = K16..23 ; lanes 16-31: +8
      v8bf b0 = *(const v8bf*)&Bl[cur][r][hi * 8];
      v8bf b1 = *(const v8bf*)&Bl[cur][r][16 + hi * 8];
      bfrag[nf] = __builtin_shufflevector(b0, b1, 0, 1, 2, 3, 4, 5, 6, 7,
                                          8, 9, 10, 11, 12, 13, 14, 15);
    }

#pragma unroll
    for (int mf = 0; mf < 2; ++mf)
#pragma unroll
      for (int nf = 0; nf < 4; ++nf)
        acc[mf][nf] = __builtin_amdgcn_wmma_f32_16x16x32_bf16(
            false, afrag[mf], false, bfrag[nf], (short)0, acc[mf][nf],
            false, false);

    __syncthreads();  // all reads of buf `cur` done before it is re-staged
  }

  // ---- epilogue: bias + activation, store bf16 ----
#pragma unroll
  for (int mf = 0; mf < 2; ++mf) {
    int gm0 = rowBase + waveM * 32 + mf * 16 + hi * 8;
#pragma unroll
    for (int nf = 0; nf < 4; ++nf) {
      int gn = colBase + waveN * 64 + nf * 16 + laneLo;
      float bb = bias ? bias[gn] : 0.f;
#pragma unroll
      for (int r = 0; r < 8; ++r) {
        float v = acc[mf][nf][r] + bb;
        if (ACT == 0) v = v > 0.f ? v : 0.f;
        else          v = tanhf(v);
        C[(size_t)(gm0 + r) * N + gn] = (bf16_t)v;
      }
    }
  }
}

// ---------------------------------------------------------------------------
extern "C" void kernel_launch(void* const* d_in, const int* in_sizes, int n_in,
                              void* d_out, int out_size, void* d_ws,
                              size_t ws_size, hipStream_t stream) {
  const int*   leaf = (const int*)d_in[0];
  const float* ent  = (const float*)d_in[1];
  const float* th   = (const float*)d_in[2];
  const float* w1   = (const float*)d_in[3];
  const float* b1   = (const float*)d_in[4];
  const float* w2   = (const float*)d_in[5];
  const float* b2   = (const float*)d_in[6];
  const float* o1   = (const float*)d_in[7];
  const float* ob1  = (const float*)d_in[8];
  const float* o2   = (const float*)d_in[9];
  const float* ob2  = (const float*)d_in[10];
  const float* h1   = (const float*)d_in[11];
  const float* hb1  = (const float*)d_in[12];
  const float* h2   = (const float*)d_in[13];
  const float* hb2  = (const float*)d_in[14];
  const float* h3   = (const float*)d_in[15];
  const float* hb3  = (const float*)d_in[16];
  const float* h4   = (const float*)d_in[17];
  const float* hb4  = (const float*)d_in[18];
  float* out = (float*)d_out;

  // ---- carve workspace (bf16 regions, 256B aligned) ----
  char* ws = (char*)d_ws;
  size_t off = 0;
  auto carve = [&](size_t halfs) -> bf16_t* {
    bf16_t* p = (bf16_t*)(ws + off);
    off += ((halfs * 2 + 255) & ~(size_t)255);
    return p;
  };
  bf16_t* w1T = carve((size_t)2048 * 1024);
  bf16_t* w2T = carve((size_t)512 * 2048);
  bf16_t* o1T = carve((size_t)2048 * 512);
  bf16_t* o2T = carve((size_t)512 * 2048);
  bf16_t* h1T = carve((size_t)512 * 1536);
  bf16_t* h2T = carve((size_t)256 * 512);
  bf16_t* h3T = carve((size_t)128 * 256);
  bf16_t* X0  = carve((size_t)65536 * 512);   // 64 MB : level-0 gather / ping
  bf16_t* Y   = carve((size_t)32768 * 512);   // 32 MB : pong
  bf16_t* H   = carve((size_t)8192 * 2048);   // 32 MB : hidden (M-chunked)
  bf16_t* feat = carve((size_t)512 * 1536);
  bf16_t* z1  = carve((size_t)512 * 512);
  bf16_t* z2  = carve((size_t)512 * 256);
  bf16_t* z3  = carve((size_t)512 * 128);

  // ---- one-time weight transposes to bf16 [N][K] ----
  auto cvt = [&](const float* in, bf16_t* o, int K, int N) {
    int n = K * N;
    cvt_transpose_kernel<<<(n + 255) / 256, 256, 0, stream>>>(in, o, K, N);
  };
  cvt(w1, w1T, 1024, 2048);
  cvt(w2, w2T, 2048, 512);
  cvt(o1, o1T, 512, 2048);
  cvt(o2, o2T, 2048, 512);
  cvt(h1, h1T, 1536, 512);
  cvt(h2, h2T, 512, 256);
  cvt(h3, h3T, 256, 128);

  // ---- gather leaves -> bf16 [65536, 512] ----
  gather_emb_kernel<<<65536, 256, 0, stream>>>(leaf, ent, X0);

  auto gemm = [&](const bf16_t* A, const bf16_t* Bt, const float* bias,
                  bf16_t* C, int M, int N, int K, int act) {
    dim3 grid(N / 128, M / 128);
    if (act == 0)
      gemm_bf16_kernel<0><<<grid, 256, 0, stream>>>(A, Bt, bias, C, M, N, K);
    else
      gemm_bf16_kernel<1><<<grid, 256, 0, stream>>>(A, Bt, bias, C, M, N, K);
  };

  // ---- 6 tree levels; reshape(2E) == reinterpret [2M,512] as [M,1024] ----
  bf16_t* inb = X0;
  bf16_t* outb = Y;
  int M = 32768;  // pair rows at level 0
  for (int lvl = 0; lvl < 6; ++lvl) {
    for (int c0 = 0; c0 < M; c0 += 8192) {
      int mc = (M - c0 < 8192) ? (M - c0) : 8192;
      gemm(inb + (size_t)c0 * 1024, w1T, b1, H, mc, 2048, 1024, 0);       // ReLU
      gemm(H, w2T, b2, outb + (size_t)c0 * 512, mc, 512, 2048, 0);       // ReLU
    }
    bf16_t* t = inb; inb = outb; outb = t;
    M >>= 1;
  }
  // after 6 swaps inb == X0, holding root [1024, 512]

  // ---- one2one: ReLU then Tanh ----
  gemm(inb, o1T, ob1, H, 1024, 2048, 512, 0);
  gemm(H, o2T, ob2, outb, 1024, 512, 2048, 1);  // tanh -> outb [1024,512]

  // ---- theorem head ----
  build_feat_kernel<<<(512 * 1536 + 255) / 256, 256, 0, stream>>>(th, outb, feat);
  gemm(feat, h1T, hb1, z1, 512, 512, 1536, 0);
  gemm(z1, h2T, hb2, z2, 512, 256, 512, 0);
  gemm(z2, h3T, hb3, z3, 512, 128, 256, 0);
  head4_kernel<<<512, 32, 0, stream>>>(z3, h4, hb4, out);
}
